// TAGCN_GRU_73589969649931
// MI455X (gfx1250) — matrix operand order repaired and checked by reading, not compiled, not run on hardware
//
#include <hip/hip_runtime.h>
#include <hip/hip_bf16.h>
#include <cstdint>

typedef __attribute__((ext_vector_type(16))) _Float16 v16h;
typedef __attribute__((ext_vector_type(8)))  float    v8f;

#define DEVI __device__ __forceinline__

constexpr int N   = 100000;   // nodes
constexpr int FS  = 32;       // static feat
constexpr int GCN = 64;       // gcn out
constexpr int EMB = 32;       // id embed
constexpr int H   = 128;      // hidden
constexpr int DYN = 16;       // dyn feat
constexpr int E   = 1600000;  // edges
constexpr int B   = 512;      // batch
constexpr int T   = 200;      // traj len
constexpr float EPSN = 1e-5f;

DEVI v8f wmma16(v16h a, v16h b, v8f c) {
  // v_wmma_f32_16x16x32_f16: D = A(16x32,f16) * B(32x16,f16) + C(f32)
  return __builtin_amdgcn_wmma_f32_16x16x32_f16(false, a, false, b, (short)0, c, false, false);
}

// B-fragment: lane holds col n = lane&15, K = (lane>>4)*8 + [0..7] and 16 + (lane>>4)*8 + [0..7]
// when weights are stored n-major.
DEVI v16h ld_b16(const _Float16* wrow) {
  v16h b;
#pragma unroll
  for (int j = 0; j < 16; ++j) b[j] = wrow[j < 8 ? j : j + 8];
  return b;
}

// ---------------- utility kernels ----------------
__global__ void k_zero_f32(float* p, int n) {
  int i = blockIdx.x * blockDim.x + threadIdx.x;
  if (i < n) p[i] = 0.f;
}

__global__ void k_cvt_f16(_Float16* dst, const float* src, int n) {
  int i = blockIdx.x * blockDim.x + threadIdx.x;
  if (i < n) dst[i] = (_Float16)src[i];
}

// proj_W [K=128, N=128] k-major -> f16 n-major [n][k]
__global__ void k_cvt_proj(_Float16* dst, const float* src) {
  int i = blockIdx.x * blockDim.x + threadIdx.x;
  if (i >= H * H) return;
  int n = i >> 7, k = i & 127;
  dst[i] = (_Float16)src[(size_t)k * H + n];
}

// tag_W [4][K=32][N=64] -> f16 [hop][n][k]
__global__ void k_cvt_tag(_Float16* dst, const float* src) {
  int i = blockIdx.x * blockDim.x + threadIdx.x;
  if (i >= 4 * FS * GCN) return;
  int hop = i >> 11;
  int r = i & 2047;
  int n = r >> 5, k = r & 31;
  dst[i] = (_Float16)src[(size_t)hop * FS * GCN + (size_t)k * GCN + n];
}

__global__ void k_init_out(float* out_acc, const float* tag_b) {
  int i = blockIdx.x * blockDim.x + threadIdx.x;
  if (i < N * GCN) out_acc[i] = tag_b[i & (GCN - 1)];
}

// ---------------- graph: degree / norm / scatter ----------------
__global__ void k_deg(float* deg, const int* col) {
  int e = blockIdx.x * blockDim.x + threadIdx.x;
  if (e < E) atomicAdd(&deg[col[e]], 1.0f);
}

__global__ void k_dinv(float* deg) {
  int i = blockIdx.x * blockDim.x + threadIdx.x;
  if (i < N) { float d = deg[i]; deg[i] = d > 0.f ? rsqrtf(d) : 0.f; }
}

// h_dst[col] += dinv[row]*dinv[col] * h_src[row]   (32 feats, 4 per thread)
__global__ void k_scatter(float* hdst, const float* hsrc,
                          const int* row, const int* col, const float* dinv) {
  int gid = blockIdx.x * blockDim.x + threadIdx.x;
  if (gid >= E * 8) return;
  int e = gid >> 3;
  int f = (gid & 7) << 2;
  int r = row[e], c = col[e];
  float w = dinv[r] * dinv[c];
  const float4 hv = *(const float4*)(hsrc + (size_t)r * FS + f);
  float* d = hdst + (size_t)c * FS + f;
  atomicAdd(d + 0, w * hv.x);
  atomicAdd(d + 1, w * hv.y);
  atomicAdd(d + 2, w * hv.z);
  atomicAdd(d + 3, w * hv.w);
}

// ---------------- out_acc[N,64] += A[N,32](f32) @ W (f16, n-major) ----------------
// one wave = one 16-row m-tile, all 4 n-tiles (A fragment reused 4x)
__global__ __launch_bounds__(256) void k_tag_gemm(float* out_acc, const float* A,
                                                  const _Float16* Wt) {
  int lane = threadIdx.x & 31, wid = threadIdx.x >> 5;
  int mt = blockIdx.x * 8 + wid;
  if (mt >= N / 16) return;
  int lm = lane & 15, kh = lane >> 4;
  const float* arow = A + (size_t)(mt * 16 + lm) * FS + kh * 8;
  v16h a;
#pragma unroll
  for (int i = 0; i < 8; ++i) {
    a[i]     = (_Float16)arow[i];
    a[8 + i] = (_Float16)arow[16 + i];
  }
#pragma unroll
  for (int nt = 0; nt < 4; ++nt) {
    int n = nt * 16 + lm;
    v16h bf = ld_b16(Wt + (size_t)n * FS + kh * 8);
    float* crow = out_acc + ((size_t)mt * 16 + kh * 8) * GCN + n;
    v8f c;
#pragma unroll
    for (int r = 0; r < 8; ++r) c[r] = crow[(size_t)r * GCN];
    c = wmma16(a, bf, c);
#pragma unroll
    for (int r = 0; r < 8; ++r) crow[(size_t)r * GCN] = c[r];
  }
}

// ---------------- BatchNorm (training stats) + ELU ----------------
__global__ void k_bn_stats(float* bnacc, const float* out_acc) {
  __shared__ float L1[256], L2[256];
  int tid = threadIdx.x;
  int c = tid & 63, g = tid >> 6;
  int base = blockIdx.x * 256;
  float s = 0.f, s2 = 0.f;
  for (int r = base + g; r < base + 256 && r < N; r += 4) {
    float v = out_acc[(size_t)r * GCN + c];
    s += v; s2 += v * v;
  }
  L1[tid] = s; L2[tid] = s2;
  __syncthreads();
  if (tid < 64) {
    s  = L1[tid] + L1[tid + 64] + L1[tid + 128] + L1[tid + 192];
    s2 = L2[tid] + L2[tid + 64] + L2[tid + 128] + L2[tid + 192];
    atomicAdd(&bnacc[tid], s);
    atomicAdd(&bnacc[64 + tid], s2);
  }
}

__global__ void k_bn_fin(float* out_acc, const float* bnacc,
                         const float* gam, const float* bet) {
  int i = blockIdx.x * blockDim.x + threadIdx.x;
  if (i >= N * GCN) return;
  int c = i & (GCN - 1);
  float mean = bnacc[c] * (1.f / (float)N);
  float var  = bnacc[64 + c] * (1.f / (float)N) - mean * mean;
  float v = gam[c] * (out_acc[i] - mean) * rsqrtf(var + EPSN) + bet[c];
  out_acc[i] = v > 0.f ? v : (expf(v) - 1.f);   // ELU(alpha=1)
}

// ---------------- gather + projection GEMM + ReLU -> xs16[T,B,H] (f16) ----------------
// 8-element chunks are 8-aligned -> never straddle the 32/96 feature boundaries.
DEVI const float* srcp(const float* id_emb, const float* gcn, const float* xst,
                       int node, int k0) {
  if (k0 < EMB)       return id_emb + (size_t)node * EMB + k0;
  if (k0 < EMB + GCN) return gcn + (size_t)node * GCN + (k0 - EMB);
  return xst + (size_t)node * FS + (k0 - EMB - GCN);
}

// one wave = one 16-row m-tile, all 8 n-tiles (gathered A fragment reused 8x)
__global__ __launch_bounds__(256) void k_proj(_Float16* xs16, const int* traj,
                                              const float* id_emb, const float* gcn,
                                              const float* xst, const _Float16* Wt,
                                              const float* bias) {
  int lane = threadIdx.x & 31, wid = threadIdx.x >> 5;
  int mt = blockIdx.x * 8 + wid;               // T*B/16 = 6400 m-tiles
  if (mt >= T * B / 16) return;
  int t = mt >> 5;                             // B/16 = 32
  int btile = mt & 31;
  int lm = lane & 15, kh = lane >> 4;
  int brow = btile * 16 + lm;
  int node = traj[(size_t)brow * T + t];
  v8f zero = {};
  v8f acc[8];
#pragma unroll
  for (int nt = 0; nt < 8; ++nt) acc[nt] = zero;
  for (int kk = 0; kk < H; kk += 32) {
    int k0 = kk + kh * 8;
    const float* p0 = srcp(id_emb, gcn, xst, node, k0);
    const float* p1 = srcp(id_emb, gcn, xst, node, k0 + 16);
    v16h a;
#pragma unroll
    for (int i = 0; i < 8; ++i) { a[i] = (_Float16)p0[i]; a[8 + i] = (_Float16)p1[i]; }
#pragma unroll
    for (int nt = 0; nt < 8; ++nt) {
      int n = nt * 16 + lm;
      v16h bf = ld_b16(Wt + (size_t)n * H + kk + kh * 8);
      acc[nt] = wmma16(a, bf, acc[nt]);
    }
  }
#pragma unroll
  for (int nt = 0; nt < 8; ++nt) {
    int n = nt * 16 + lm;
    float bn = bias[n];
#pragma unroll
    for (int r = 0; r < 8; ++r) {
      float y = acc[nt][r] + bn;
      y = y > 0.f ? y : 0.f;                   // ReLU
      int rowb = btile * 16 + kh * 8 + r;
      xs16[((size_t)t * B + rowb) * H + n] = (_Float16)y;
    }
  }
}

// ---------------- fused GRU step: gate GEMMs in LDS + masked update ----------------
// block = (m-tile, dir): owns 16 batch rows of one direction.
// waves 0-3 compute Gi (x_t @ Wih^T), waves 4-7 compute Gh (h @ Whh^T), each wave
// handles 6 of the 24 n-tiles with the A fragment reused across them. Gh's A rows
// are exactly this block's rows, so the post-barrier h-update has no cross-block
// hazard. Gates staged in LDS (2 x 16 x 384 f32 = 48 KB of the 320 KB WGP pool).
__global__ __launch_bounds__(256) void k_gru_step(
    float* hstate, _Float16* h16, const _Float16* xs16, const _Float16* w16gru,
    const float* bih_f, const float* bhh_f, const float* bih_b, const float* bhh_b,
    const int* lens, int step) {
  __shared__ float ldsG[2][16][3 * H];
  int lane = threadIdx.x & 31, wid = threadIdx.x >> 5;
  int mt = blockIdx.x;                         // 0..31
  int dir = blockIdx.y;                        // 0..1
  int t = dir ? (T - 1 - step) : step;
  int lm = lane & 15, kh = lane >> 4;
  int isH = wid >> 2;
  int ntb = wid & 3;
  const _Float16* A = isH ? (h16 + (size_t)dir * B * H)
                          : (xs16 + (size_t)t * B * H);
  const _Float16* W = w16gru + (size_t)(dir * 2 + isH) * (3 * H * H);
  const float* bias = isH ? (dir ? bhh_b : bhh_f) : (dir ? bih_b : bih_f);
  const _Float16* arow = A + (size_t)(mt * 16 + lm) * H;
  v8f zero = {};
  v8f acc[6];
#pragma unroll
  for (int i = 0; i < 6; ++i) acc[i] = zero;
  for (int kk = 0; kk < H; kk += 32) {
    int kb = kk + kh * 8;
    v16h a;
#pragma unroll
    for (int i = 0; i < 8; ++i) { a[i] = arow[kb + i]; a[8 + i] = arow[kb + 16 + i]; }
#pragma unroll
    for (int i = 0; i < 6; ++i) {
      int n = (ntb + 4 * i) * 16 + lm;
      v16h bf = ld_b16(W + (size_t)n * H + kk + kh * 8);
      acc[i] = wmma16(a, bf, acc[i]);
    }
  }
#pragma unroll
  for (int i = 0; i < 6; ++i) {
    int n = (ntb + 4 * i) * 16 + lm;
    float bn = bias[n];
#pragma unroll
    for (int r = 0; r < 8; ++r) ldsG[isH][kh * 8 + r][n] = acc[i][r] + bn;
  }
  __syncthreads();
  // update: 16 rows x 128 cols, 8 consecutive j per thread
  int row = threadIdx.x >> 4;
  int j0 = (threadIdx.x & 15) * 8;
  int b = mt * 16 + row;
  bool msk = t < lens[b];
  size_t hbase = (size_t)dir * B * H + (size_t)b * H;
#pragma unroll
  for (int q = 0; q < 8; ++q) {
    int j = j0 + q;
    float ir = ldsG[0][row][j], iz = ldsG[0][row][H + j], inn = ldsG[0][row][2 * H + j];
    float hr = ldsG[1][row][j], hz = ldsG[1][row][H + j], hn  = ldsG[1][row][2 * H + j];
    float hp = hstate[hbase + j];
    float r = 1.f / (1.f + expf(-(ir + hr)));
    float z = 1.f / (1.f + expf(-(iz + hz)));
    float nn = tanhf(inn + r * hn);
    float hnew = (1.f - z) * nn + z * hp;
    float h = msk ? hnew : hp;
    hstate[hbase + j] = h;
    h16[hbase + j] = (_Float16)h;
  }
}

// ---------------- head: LayerNorm + fc1 + GELU + fc2 (one block per batch row) ------------
__global__ void k_head(float* out, const float* hstate, const float* dyn,
                       const float* lng, const float* lnb,
                       const float* fc1W, const float* fc1b,
                       const float* fc2W, const float* fc2b) {
  __shared__ float sbuf[2 * H + DYN];   // 272
  __shared__ float red[256];
  int b = blockIdx.x, tid = threadIdx.x;
  float s = (tid < H) ? hstate[(size_t)b * H + tid]
                      : hstate[(size_t)B * H + (size_t)b * H + (tid - H)];
  red[tid] = s;
  __syncthreads();
  for (int off = 128; off > 0; off >>= 1) {
    if (tid < off) red[tid] += red[tid + off];
    __syncthreads();
  }
  float mean = red[0] * (1.f / 256.f);
  __syncthreads();
  red[tid] = s * s;
  __syncthreads();
  for (int off = 128; off > 0; off >>= 1) {
    if (tid < off) red[tid] += red[tid + off];
    __syncthreads();
  }
  float var = red[0] * (1.f / 256.f) - mean * mean;
  __syncthreads();
  sbuf[tid] = lng[tid] * (s - mean) * rsqrtf(var + EPSN) + lnb[tid];
  if (tid < DYN) sbuf[2 * H + tid] = dyn[(size_t)b * DYN + tid];
  __syncthreads();
  float g = 0.f;
  if (tid < H) {
    float acc = fc1b[tid];
    for (int k = 0; k < 2 * H + DYN; ++k) acc += sbuf[k] * fc1W[(size_t)k * H + tid];
    float ge = 0.5f * acc * (1.f + erff(acc * 0.7071067811865475f));  // exact GELU
    g = ge * fc2W[tid];
  }
  __syncthreads();
  red[tid] = (tid < H) ? g : 0.f;
  __syncthreads();
  for (int off = 128; off > 0; off >>= 1) {
    if (tid < off) red[tid] += red[tid + off];
    __syncthreads();
  }
  if (tid == 0) out[b] = red[0] + fc2b[0];
}

// ---------------- host launcher ----------------
extern "C" void kernel_launch(void* const* d_in, const int* in_sizes, int n_in,
                              void* d_out, int out_size, void* d_ws, size_t ws_size,
                              hipStream_t stream) {
  (void)in_sizes; (void)n_in; (void)out_size; (void)ws_size;
  const float* x_static = (const float*)d_in[0];
  const int*   edge     = (const int*)d_in[1];     // [2,E]: row = edge, col = edge+E
  const int*   traj     = (const int*)d_in[2];
  const int*   lens     = (const int*)d_in[3];
  const float* dyn      = (const float*)d_in[4];
  const float* id_emb   = (const float*)d_in[5];
  const float* tag_W    = (const float*)d_in[6];
  const float* tag_b    = (const float*)d_in[7];
  const float* bn_g     = (const float*)d_in[8];
  const float* bn_b     = (const float*)d_in[9];
  const float* proj_W   = (const float*)d_in[10];
  const float* proj_b   = (const float*)d_in[11];
  const float* Wih_f    = (const float*)d_in[12];
  const float* Whh_f    = (const float*)d_in[13];
  const float* bih_f    = (const float*)d_in[14];
  const float* bhh_f    = (const float*)d_in[15];
  const float* Wih_b    = (const float*)d_in[16];
  const float* Whh_b    = (const float*)d_in[17];
  const float* bih_b    = (const float*)d_in[18];
  const float* bhh_b    = (const float*)d_in[19];
  const float* ln_g     = (const float*)d_in[20];
  const float* ln_b     = (const float*)d_in[21];
  const float* fc1W     = (const float*)d_in[22];
  const float* fc1b     = (const float*)d_in[23];
  const float* fc2W     = (const float*)d_in[24];
  const float* fc2b     = (const float*)d_in[25];
  float* outp = (float*)d_out;

  // workspace carve-up (256B aligned)
  char* p = (char*)d_ws;
  auto alloc = [&](size_t bytes) { char* r = p; p += (bytes + 255) & ~(size_t)255; return r; };
  float*    deg    = (float*)alloc((size_t)N * 4);
  float*    hA     = (float*)alloc((size_t)N * FS * 4);
  float*    hB     = (float*)alloc((size_t)N * FS * 4);
  float*    gout   = (float*)alloc((size_t)N * GCN * 4);           // TAG out -> gcn (in place)
  float*    bnacc  = (float*)alloc(128 * 4);
  _Float16* xs16   = (_Float16*)alloc((size_t)T * B * H * 2);
  float*    hstate = (float*)alloc((size_t)2 * B * H * 4);
  _Float16* h16    = (_Float16*)alloc((size_t)2 * B * H * 2);
  _Float16* w16    = (_Float16*)alloc((size_t)(H * H + 4 * 3 * H * H + 4 * FS * GCN) * 2);
  _Float16* w16_proj = w16;                    // [n][k] 128x128
  _Float16* w16_gru  = w16 + H * H;            // 4 x [3H][H], job order
  _Float16* w16_tag  = w16_gru + 4 * 3 * H * H; // [hop][n][k] 4x64x32

  const int TB = 256;
  // weight conversions to f16 (proj/tag transposed to n-major)
  k_cvt_proj<<<(H * H + TB - 1) / TB, TB, 0, stream>>>(w16_proj, proj_W);
  k_cvt_f16<<<(3 * H * H + TB - 1) / TB, TB, 0, stream>>>(w16_gru + 0 * 3 * H * H, Wih_f, 3 * H * H);
  k_cvt_f16<<<(3 * H * H + TB - 1) / TB, TB, 0, stream>>>(w16_gru + 1 * 3 * H * H, Whh_f, 3 * H * H);
  k_cvt_f16<<<(3 * H * H + TB - 1) / TB, TB, 0, stream>>>(w16_gru + 2 * 3 * H * H, Wih_b, 3 * H * H);
  k_cvt_f16<<<(3 * H * H + TB - 1) / TB, TB, 0, stream>>>(w16_gru + 3 * 3 * H * H, Whh_b, 3 * H * H);
  k_cvt_tag<<<(4 * FS * GCN + TB - 1) / TB, TB, 0, stream>>>(w16_tag, tag_W);

  // zero init
  k_zero_f32<<<(N + TB - 1) / TB, TB, 0, stream>>>(deg, N);
  k_zero_f32<<<1, 128, 0, stream>>>(bnacc, 128);
  k_zero_f32<<<(2 * B * H + TB - 1) / TB, TB, 0, stream>>>(hstate, 2 * B * H);
  k_zero_f32<<<(B * H + TB - 1) / TB, TB, 0, stream>>>((float*)h16, B * H); // 2*B*H halves
  k_init_out<<<(N * GCN + TB - 1) / TB, TB, 0, stream>>>(gout, tag_b);

  // graph norm
  k_deg<<<(E + TB - 1) / TB, TB, 0, stream>>>(deg, edge + E);
  k_dinv<<<(N + TB - 1) / TB, TB, 0, stream>>>(deg);

  const int TAG_BLOCKS = (N / 16 + 7) / 8;   // 782
  // hop 0: out += x_static @ W0
  k_tag_gemm<<<TAG_BLOCKS, TB, 0, stream>>>(gout, x_static, w16_tag + 0 * FS * GCN);
  // hop 1
  k_zero_f32<<<(N * FS + TB - 1) / TB, TB, 0, stream>>>(hB, N * FS);
  k_scatter<<<(E * 8 + TB - 1) / TB, TB, 0, stream>>>(hB, x_static, edge, edge + E, deg);
  k_tag_gemm<<<TAG_BLOCKS, TB, 0, stream>>>(gout, hB, w16_tag + 1 * FS * GCN);
  // hop 2
  k_zero_f32<<<(N * FS + TB - 1) / TB, TB, 0, stream>>>(hA, N * FS);
  k_scatter<<<(E * 8 + TB - 1) / TB, TB, 0, stream>>>(hA, hB, edge, edge + E, deg);
  k_tag_gemm<<<TAG_BLOCKS, TB, 0, stream>>>(gout, hA, w16_tag + 2 * FS * GCN);
  // hop 3
  k_zero_f32<<<(N * FS + TB - 1) / TB, TB, 0, stream>>>(hB, N * FS);
  k_scatter<<<(E * 8 + TB - 1) / TB, TB, 0, stream>>>(hB, hA, edge, edge + E, deg);
  k_tag_gemm<<<TAG_BLOCKS, TB, 0, stream>>>(gout, hB, w16_tag + 3 * FS * GCN);

  // batchnorm + ELU
  k_bn_stats<<<(N + 255) / 256, 256, 0, stream>>>(bnacc, gout);
  k_bn_fin<<<(N * GCN + TB - 1) / TB, TB, 0, stream>>>(gout, bnacc, bn_g, bn_b);

  // gather + projection -> xs16
  k_proj<<<(T * B / 16) / 8, TB, 0, stream>>>(xs16, traj, id_emb, gout, x_static,
                                              w16_proj, proj_b);

  // bidirectional GRU: one fused kernel per step (gates in LDS, no global staging)
  for (int s = 0; s < T; ++s) {
    k_gru_step<<<dim3(B / 16, 2), TB, 0, stream>>>(hstate, h16, xs16, w16_gru,
                                                   bih_f, bhh_f, bih_b, bhh_b, lens, s);
  }

  // LN + fc1 + GELU + fc2
  k_head<<<B, 256, 0, stream>>>(outp, hstate, dyn, ln_g, ln_b, fc1W, fc1b, fc2W, fc2b);
}
